// JODODiffusion_42588895707438
// MI455X (gfx1250) — matrix operand order repaired and checked by reading, not compiled, not run on hardware
//
#include <hip/hip_runtime.h>
#include <hip/hip_bf16.h>

typedef __attribute__((ext_vector_type(16))) _Float16 v16h;
typedef __attribute__((ext_vector_type(8)))  float    v8f;

#define NEDGE  320000
// fragment buffer element counts (halves)
#define FT_N   131072   /* W_time: 32 nt * 8 kt * 512, tile = nt*8+kt */
#define FI_N   180224   /* W_in : 22 kt * 16 nt * 512, tile = kt*16+nt */
#define FC1_N  65536    /* W_c1 :  8 kt * 16 nt * 512, tile = kt*16+nt */

#if defined(__has_builtin)
#if __has_builtin(__builtin_amdgcn_global_load_async_to_lds_b128)
#define HAS_ASYNC 1
#endif
#endif
#ifndef HAS_ASYNC
#define HAS_ASYNC 0
#endif

// exact pointee type from the hipcc diagnostic: 'int __vector_size__(4*sizeof(int))'
typedef int v4i_vs __attribute__((__vector_size__(16)));
typedef __attribute__((address_space(1))) v4i_vs* as1_v4i;
typedef __attribute__((address_space(3))) v4i_vs* as3_v4i;

union V16 { v16h h; uint4 q[2]; };
union H8  { _Float16 h[8]; uint4 q; };

__device__ __forceinline__ float silu_f(float x) { return x / (1.0f + expf(-x)); }

// ---- 16B global->LDS copy: async-to-LDS if available, else load+ds_store ----
__device__ __forceinline__ void cp16(const _Float16* __restrict__ g, _Float16* l) {
#if HAS_ASYNC
  __builtin_amdgcn_global_load_async_to_lds_b128((as1_v4i)g, (as3_v4i)l, 0, 0);
#else
  *(uint4*)l = *(const uint4*)g;
#endif
}

__device__ __forceinline__ void wait_async() {
#if HAS_ASYNC
#if defined(__has_builtin) && __has_builtin(__builtin_amdgcn_s_wait_asynccnt)
  __builtin_amdgcn_s_wait_asynccnt(0);
#else
  asm volatile("s_wait_asynccnt 0x0" ::: "memory");
#endif
#endif
}

// cooperative copy of nhalves (multiple of 1024) by 128 threads
__device__ __forceinline__ void stage(const _Float16* __restrict__ g,
                                      _Float16* l, int nhalves) {
  for (int i = threadIdx.x * 8; i < nhalves; i += 128 * 8)
    cp16(g + i, l + i);
}

// Build a 16x32 f16 A-fragment for this lane from two contiguous 8-float runs.
__device__ __forceinline__ v16h pack_frag(const float* __restrict__ r0,
                                          const float* __restrict__ r1) {
  const float4 a0 = *(const float4*)(r0);
  const float4 a1 = *(const float4*)(r0 + 4);
  const float4 b0 = *(const float4*)(r1);
  const float4 b1 = *(const float4*)(r1 + 4);
  v16h v;
  v[0]=(_Float16)a0.x;  v[1]=(_Float16)a0.y;  v[2]=(_Float16)a0.z;  v[3]=(_Float16)a0.w;
  v[4]=(_Float16)a1.x;  v[5]=(_Float16)a1.y;  v[6]=(_Float16)a1.z;  v[7]=(_Float16)a1.w;
  v[8]=(_Float16)b0.x;  v[9]=(_Float16)b0.y;  v[10]=(_Float16)b0.z; v[11]=(_Float16)b0.w;
  v[12]=(_Float16)b1.x; v[13]=(_Float16)b1.y; v[14]=(_Float16)b1.z; v[15]=(_Float16)b1.w;
  return v;
}

// Source pointer into the virtual 704-wide concat [h[row] | h[col] | edge_attr | dist]
__device__ __forceinline__ const float* hin_src(int kr, const float* __restrict__ h,
                                                const float* __restrict__ ea,
                                                const float* __restrict__ di,
                                                long long ri, long long ci, long long e) {
  if (kr < 256) return h + ri * 256 + kr;
  if (kr < 512) return h + ci * 256 + (kr - 256);
  if (kr < 640) return ea + e * 128 + (kr - 512);
  return di + e * 64 + (kr - 640);
}

// ---------------- prep: f32 weights -> f16 WMMA-B-fragment order in d_ws ----
__global__ void prep_weights(const float* __restrict__ Wt,
                             const float* __restrict__ Wi,
                             const float* __restrict__ Wc1,
                             _Float16* __restrict__ FT,
                             _Float16* __restrict__ FI,
                             _Float16* __restrict__ FC1) {
  int idx = blockIdx.x * blockDim.x + threadIdx.x;
  if (idx >= FT_N + FI_N + FC1_N) return;
  // B-fragment (32x16 f16): lane%16 = n, lanes 0-15 hold K 0..7,16..23; 16-31 hold 8..15,24..31
  if (idx < FT_N) {                      // tile = nt*8 + kt  (fused-group contiguous)
    int v = idx & 15, lane = (idx >> 4) & 31, tile = idx >> 9;
    int kt = tile & 7, nt = tile >> 3;
    int k = kt * 32 + ((v >> 3) << 4) + ((lane >> 4) << 3) + (v & 7);
    int n = nt * 16 + (lane & 15);
    FT[idx] = (_Float16)Wt[k * 512 + n];
  } else if (idx < FT_N + FI_N) {        // tile = kt*16 + nt
    int j = idx - FT_N;
    int v = j & 15, lane = (j >> 4) & 31, tile = j >> 9;
    int nt = tile & 15, kt = tile >> 4;
    int k = kt * 32 + ((v >> 3) << 4) + ((lane >> 4) << 3) + (v & 7);
    int n = nt * 16 + (lane & 15);
    FI[j] = (_Float16)Wi[k * 256 + n];
  } else {                               // tile = kt*16 + nt
    int j = idx - FT_N - FI_N;
    int v = j & 15, lane = (j >> 4) & 31, tile = j >> 9;
    int nt = tile & 15, kt = tile >> 4;
    int k = kt * 32 + ((v >> 3) << 4) + ((lane >> 4) << 3) + (v & 7);
    int n = nt * 16 + (lane & 15);
    FC1[j] = (_Float16)Wc1[k * 256 + n];
  }
}

__global__ void init_out(const float* __restrict__ pos, float* __restrict__ out, int n) {
  int i = blockIdx.x * blockDim.x + threadIdx.x;
  if (i < n) out[i] = pos[i];
}

// -- main: 4 waves/block, 16 edges/wave, double-buffered async B staging -----
__global__ void __launch_bounds__(128)
egnn_edge_kernel(const float* __restrict__ h,
                 const float* __restrict__ pos,
                 const float* __restrict__ edge_attr,
                 const float* __restrict__ dist,
                 const float* __restrict__ time_emb,
                 const long long* __restrict__ eidx,
                 const float* __restrict__ b_time,
                 const float* __restrict__ b_in,
                 const float* __restrict__ b_c1,
                 const float* __restrict__ Wc2,
                 const float* __restrict__ coord_scale,
                 const _Float16* __restrict__ FT,
                 const _Float16* __restrict__ FI,
                 const _Float16* __restrict__ FC1,
                 float* __restrict__ out) {
  const int tid  = threadIdx.x;
  const int wave = tid >> 5;
  const int lane = tid & 31;
  const int m    = lane & 15;
  const int hf   = lane >> 4;
  const long long e = ((long long)blockIdx.x * 4 + wave) * 16 + m;

  const long long ri = eidx[e];
  const long long ci = eidx[(long long)NEDGE + e];

  __shared__ _Float16 sB[2][8192];        // 2 x 16KB: double-buffered B frags
  __shared__ _Float16 sAt[4][16 * 264];   // per-wave silu(time_emb) f16
  __shared__ _Float16 sI[4][16 * 264];    // per-wave modulated inv f16
  __shared__ float    sS[4][16];          // per-wave row scalars

  // ---- prologue: kick off first B group, overlap with silu(time_emb) ------
  stage(FI, sB[0], 8192);
  {
    const float* te = time_emb + e * 256;
    for (int c0 = 0; c0 < 128; c0 += 8) {
      const int col = hf * 128 + c0;
      const float4 x0 = *(const float4*)(te + col);
      const float4 x1 = *(const float4*)(te + col + 4);
      H8 u;
      u.h[0]=(_Float16)silu_f(x0.x); u.h[1]=(_Float16)silu_f(x0.y);
      u.h[2]=(_Float16)silu_f(x0.z); u.h[3]=(_Float16)silu_f(x0.w);
      u.h[4]=(_Float16)silu_f(x1.x); u.h[5]=(_Float16)silu_f(x1.y);
      u.h[6]=(_Float16)silu_f(x1.z); u.h[7]=(_Float16)silu_f(x1.w);
      *(uint4*)&sAt[wave][m * 264 + col] = u.q;
    }
  }
  wait_async();
  __syncthreads();

  // ---- Phase A: raw = h_input @ W_in (+b_in), acc in regs, pipelined B ----
  v8f acc[16];
  #pragma unroll
  for (int nt = 0; nt < 16; ++nt) { v8f z = {}; acc[nt] = z; }
  for (int kt = 0; kt < 22; ++kt) {
    const _Float16* cur = sB[kt & 1];
    if (kt + 1 < 22) stage(FI + (size_t)(kt + 1) * 8192, sB[(kt + 1) & 1], 8192);
    const int run0 = kt * 32 + hf * 8;
    v16h a = pack_frag(hin_src(run0,      h, edge_attr, dist, ri, ci, e),
                       hin_src(run0 + 16, h, edge_attr, dist, ri, ci, e));
    #pragma unroll
    for (int nt = 0; nt < 16; ++nt) {
      v16h b = *(const v16h*)(cur + ((nt * 32 + lane) << 4));
      acc[nt] = __builtin_amdgcn_wmma_f32_16x16x32_f16(false, a, false, b, (short)0, acc[nt], false, false);
    }
    wait_async();
    __syncthreads();
  }

  // kick off Phase B's first group, overlap with in-register LayerNorm ------
  stage(FT,                      sB[0],        4096);
  stage(FT + (size_t)16 * 4096,  sB[0] + 4096, 4096);
  float sum[8], sq[8];
  #pragma unroll
  for (int g = 0; g < 8; ++g) { sum[g] = 0.f; sq[g] = 0.f; }
  #pragma unroll
  for (int nt = 0; nt < 16; ++nt) {
    const float bb = b_in[nt * 16 + m];
    #pragma unroll
    for (int g = 0; g < 8; ++g) {
      const float x = acc[nt][g] + bb;
      acc[nt][g] = x;
      sum[g] += x; sq[g] += x * x;
    }
  }
  #pragma unroll
  for (int mask = 1; mask < 16; mask <<= 1) {
    #pragma unroll
    for (int g = 0; g < 8; ++g) {
      sum[g] += __shfl_xor(sum[g], mask, 32);
      sq[g]  += __shfl_xor(sq[g],  mask, 32);
    }
  }
  float mu[8], rs[8];
  #pragma unroll
  for (int g = 0; g < 8; ++g) {
    mu[g] = sum[g] * (1.0f / 256.0f);
    const float var = sq[g] * (1.0f / 256.0f) - mu[g] * mu[g];
    rs[g] = rsqrtf(var + 1e-6f);
  }
  wait_async();
  __syncthreads();

  // ---- Phase B (fused): t tiles (shift/scale) + AdaLN modulate -> sI ------
  #pragma unroll
  for (int nt = 0; nt < 16; ++nt) {
    const _Float16* cur = sB[nt & 1];
    if (nt + 1 < 16) {
      stage(FT + (size_t)(nt + 1)  * 4096, sB[(nt + 1) & 1],        4096);
      stage(FT + (size_t)(nt + 17) * 4096, sB[(nt + 1) & 1] + 4096, 4096);
    }
    v8f csh = {}, csc = {};
    #pragma unroll
    for (int kt = 0; kt < 8; ++kt) {
      const int base = kt * 32 + hf * 8;
      V16 av;
      av.q[0] = *(const uint4*)&sAt[wave][m * 264 + base];
      av.q[1] = *(const uint4*)&sAt[wave][m * 264 + base + 16];
      v16h b0 = *(const v16h*)(cur +        ((kt * 32 + lane) << 4));
      v16h b1 = *(const v16h*)(cur + 4096 + ((kt * 32 + lane) << 4));
      csh = __builtin_amdgcn_wmma_f32_16x16x32_f16(false, av.h, false, b0, (short)0, csh, false, false);
      csc = __builtin_amdgcn_wmma_f32_16x16x32_f16(false, av.h, false, b1, (short)0, csc, false, false);
    }
    const int col = nt * 16 + m;
    const float bsh = b_time[col];
    const float bsc = b_time[256 + col];
    #pragma unroll
    for (int g = 0; g < 8; ++g) {
      const float shift = csh[g] + bsh;
      const float scale = csc[g] + bsc;
      const float xh = (acc[nt][g] - mu[g]) * rs[g];
      sI[wave][(g + hf * 8) * 264 + col] = (_Float16)(xh * (1.0f + scale) + shift);
    }
    wait_async();
    __syncthreads();
  }

  // ---- Phase C: c1 GEMM + silu + dot(W_c2) in registers -------------------
  stage(FC1, sB[0], 8192);
  wait_async();
  __syncthreads();
  v8f ac2[16];
  #pragma unroll
  for (int nt = 0; nt < 16; ++nt) { v8f z = {}; ac2[nt] = z; }
  for (int kt = 0; kt < 8; ++kt) {
    const _Float16* cur = sB[kt & 1];
    if (kt + 1 < 8) stage(FC1 + (size_t)(kt + 1) * 8192, sB[(kt + 1) & 1], 8192);
    const int base = kt * 32 + hf * 8;
    V16 av;
    av.q[0] = *(const uint4*)&sI[wave][m * 264 + base];
    av.q[1] = *(const uint4*)&sI[wave][m * 264 + base + 16];
    #pragma unroll
    for (int nt = 0; nt < 16; ++nt) {
      v16h b = *(const v16h*)(cur + ((nt * 32 + lane) << 4));
      ac2[nt] = __builtin_amdgcn_wmma_f32_16x16x32_f16(false, av.h, false, b, (short)0, ac2[nt], false, false);
    }
    wait_async();
    __syncthreads();
  }
  float part[8];
  #pragma unroll
  for (int g = 0; g < 8; ++g) part[g] = 0.f;
  #pragma unroll
  for (int nt = 0; nt < 16; ++nt) {
    const int col = nt * 16 + m;
    const float bb = b_c1[col];
    const float w  = Wc2[col];
    #pragma unroll
    for (int g = 0; g < 8; ++g)
      part[g] += silu_f(ac2[nt][g] + bb) * w;
  }
  #pragma unroll
  for (int mask = 1; mask < 16; mask <<= 1) {
    #pragma unroll
    for (int g = 0; g < 8; ++g)
      part[g] += __shfl_xor(part[g], mask, 32);
  }
  if (m == 0) {
    #pragma unroll
    for (int g = 0; g < 8; ++g)
      sS[wave][g + hf * 8] = part[g];
  }
  __syncthreads();

  // ---- Phase D: coords + scatter-add --------------------------------------
  if (lane < 16) {
    const float inv_s = tanhf(sS[wave][lane]);
    const float* pr = pos + ri * 3;
    const float* pc = pos + ci * 3;
    const float dx = pr[0] - pc[0];
    const float dy = pr[1] - pc[1];
    const float dz = pr[2] - pc[2];
    const float nrm = sqrtf(dx * dx + dy * dy + dz * dz);
    const float f = coord_scale[0] / fmaxf(nrm, 1e-8f) * inv_s;
    atomicAdd(out + ri * 3 + 0, dx * f);
    atomicAdd(out + ri * 3 + 1, dy * f);
    atomicAdd(out + ri * 3 + 2, dz * f);
  }
}

extern "C" void kernel_launch(void* const* d_in, const int* in_sizes, int n_in,
                              void* d_out, int out_size, void* d_ws, size_t ws_size,
                              hipStream_t stream) {
  const float*     h   = (const float*)d_in[0];
  const float*     pos = (const float*)d_in[1];
  const float*     ea  = (const float*)d_in[2];
  const float*     di  = (const float*)d_in[3];
  const float*     te  = (const float*)d_in[4];
  const long long* ei  = (const long long*)d_in[5];
  const float*     Wt  = (const float*)d_in[6];
  const float*     bt  = (const float*)d_in[7];
  const float*     Wi  = (const float*)d_in[8];
  const float*     bi  = (const float*)d_in[9];
  const float*     Wc1 = (const float*)d_in[10];
  const float*     bc1 = (const float*)d_in[11];
  const float*     Wc2 = (const float*)d_in[12];
  const float*     cs  = (const float*)d_in[13];
  float* out = (float*)d_out;

  _Float16* FT  = (_Float16*)d_ws;
  _Float16* FI  = FT + FT_N;
  _Float16* FC1 = FI + FI_N;

  const int prep_total = FT_N + FI_N + FC1_N;
  prep_weights<<<(prep_total + 255) / 256, 256, 0, stream>>>(Wt, Wi, Wc1, FT, FI, FC1);
  init_out<<<(out_size + 255) / 256, 256, 0, stream>>>(pos, out, out_size);
  egnn_edge_kernel<<<NEDGE / 64, 128, 0, stream>>>(h, pos, ea, di, te, ei,
                                                   bt, bi, bc1, Wc2, cs,
                                                   FT, FI, FC1, out);
}
